// GridSmoother_27118423506990
// MI455X (gfx1250) — matrix operand order repaired and checked by reading, not compiled, not run on hardware
//
#include <hip/hip_runtime.h>
#include <hip/hip_bf16.h>
#include <math.h>

constexpr int cB = 16, cP = 8192, cN = 1024, cD = 384, cL = 12, cH = 6, cHD = 64;
constexpr int cM  = cB * cN;   // 16384 total rows
constexpr int c3D = 3 * cD;    // 1152
constexpr int c4D = 4 * cD;    // 1536

typedef __attribute__((ext_vector_type(16))) __bf16 v16bf;
typedef __attribute__((ext_vector_type(8)))  float  v8f;

union Frag { v16bf v; unsigned int u[8]; unsigned short s[16]; };

__device__ __forceinline__ unsigned short f2bf(float f) {
  unsigned u = __float_as_uint(f);
  u += 0x7FFFu + ((u >> 16) & 1u);
  return (unsigned short)(u >> 16);
}
__device__ __forceinline__ float bf2f(unsigned short h) {
  return __uint_as_float(((unsigned)h) << 16);
}
__device__ __forceinline__ v8f vzero() {
  v8f x;
#pragma unroll
  for (int i = 0; i < 8; ++i) x[i] = 0.f;
  return x;
}
__device__ __forceinline__ v8f wmma_bf16(const Frag& a, const Frag& b, v8f c) {
  return __builtin_amdgcn_wmma_f32_16x16x32_bf16(false, a.v, false, b.v, (short)0, c, false, false);
}
// A-fragment (16x32 bf16) k index for packed-u32 slot jj, contiguous order:
// VGPR0-7: K = 16*half + {0..15}
__device__ __forceinline__ int akk(int jj, int half) {
  return 16 * half + 2 * jj;
}
__device__ __forceinline__ float rmax16(float v) {
#pragma unroll
  for (int m = 1; m < 16; m <<= 1) v = fmaxf(v, __shfl_xor(v, m, 32));
  return v;
}
__device__ __forceinline__ float rsum16(float v) {
#pragma unroll
  for (int m = 1; m < 16; m <<= 1) v += __shfl_xor(v, m, 32);
  return v;
}

// ---------------- elementwise helpers ----------------
__global__ void k_cvt(const float* __restrict__ src, unsigned short* __restrict__ dst, int n) {
  int i = blockIdx.x * blockDim.x + threadIdx.x;
  if (i < n) dst[i] = f2bf(src[i]);
}

__global__ void k_embed(const float* __restrict__ grid, const float* __restrict__ ew,
                        float* __restrict__ xf) {
  int idx = blockIdx.x * blockDim.x + threadIdx.x;
  if (idx >= cM * cD) return;
  int d = idx % cD, bn = idx / cD;
  xf[idx] = grid[bn * 3] * ew[d] + grid[bn * 3 + 1] * ew[cD + d] + grid[bn * 3 + 2] * ew[2 * cD + d];
}

__global__ __launch_bounds__(256)
void k_layernorm(const float* __restrict__ x, const float* __restrict__ w,
                 const float* __restrict__ bvec, unsigned short* __restrict__ out) {
  int row  = blockIdx.x * 8 + (threadIdx.x >> 5);
  int lane = threadIdx.x & 31;
  const float* xr = x + (size_t)row * cD;
  float vals[12];
  float s = 0.f, s2 = 0.f;
#pragma unroll
  for (int i = 0; i < 12; ++i) {
    float v = xr[lane + 32 * i];
    vals[i] = v; s += v; s2 += v * v;
  }
#pragma unroll
  for (int m = 1; m < 32; m <<= 1) { s += __shfl_xor(s, m, 32); s2 += __shfl_xor(s2, m, 32); }
  float mean = s / cD;
  float var  = s2 / cD - mean * mean;
  float rs   = rsqrtf(var + 1e-5f);
  unsigned short* orow = out + (size_t)row * cD;
#pragma unroll
  for (int i = 0; i < 12; ++i) {
    int d = lane + 32 * i;
    orow[d] = f2bf((vals[i] - mean) * rs * w[d] + bvec[d]);
  }
}

// ---------------- WMMA GEMM: C = A(bf16)[M,K] * W(bf16)[K,N] + bias (+epilogue) ----------------
// OUTMODE 0: store bf16.  OUTMODE 1: f32 residual accumulate (out += result).
template <int OUTMODE, bool GELU>
__global__ __launch_bounds__(128)
void k_gemm(const unsigned short* __restrict__ A, const unsigned short* __restrict__ W,
            const float* __restrict__ bias, void* __restrict__ outv,
            int M, int K, int Nc) {
  __shared__ unsigned short lA[64 * 36];   // [row][k], stride 36
  __shared__ unsigned short lB[64 * 36];   // transposed: [n][k], stride 36
  const int tid = threadIdx.x, wave = tid >> 5, lane = tid & 31;
  const int half = lane >> 4, l15 = lane & 15;
  const int m0 = blockIdx.y * 64, n0 = blockIdx.x * 64;
  const unsigned int* A32 = (const unsigned int*)A;
  const unsigned int* W32 = (const unsigned int*)W;

  v8f acc[4];
#pragma unroll
  for (int t = 0; t < 4; ++t) acc[t] = vzero();

  for (int k0 = 0; k0 < K; k0 += 32) {
#pragma unroll
    for (int j = 0; j < 8; ++j) {                 // A tile 64x32: 1024 u32
      int idx = tid + 128 * j;
      int row = idx >> 4, cp = idx & 15;
      unsigned int v = A32[((size_t)(m0 + row) * K + k0 + 2 * cp) >> 1];
      *(unsigned int*)&lA[row * 36 + 2 * cp] = v;
    }
#pragma unroll
    for (int j = 0; j < 8; ++j) {                 // W tile 32x64, store transposed
      int idx = tid + 128 * j;
      int kr = idx >> 5, np = idx & 31;
      unsigned int v = W32[((size_t)(k0 + kr) * Nc + n0 + 2 * np) >> 1];
      lB[(2 * np) * 36 + kr]     = (unsigned short)(v & 0xFFFFu);
      lB[(2 * np + 1) * 36 + kr] = (unsigned short)(v >> 16);
    }
    if (k0 + 32 < K)
      __builtin_prefetch(&A[(size_t)(m0 + (tid >> 1)) * K + k0 + 32], 0, 1);
    __syncthreads();

    Frag a;
#pragma unroll
    for (int j = 0; j < 8; ++j)
      a.u[j] = *(const unsigned int*)&lA[(wave * 16 + l15) * 36 + akk(j, half)];
#pragma unroll
    for (int t = 0; t < 4; ++t) {
      Frag b;
#pragma unroll
      for (int j = 0; j < 8; ++j)
        b.u[j] = *(const unsigned int*)&lB[(16 * t + l15) * 36 + 16 * half + 2 * j];
      acc[t] = wmma_bf16(a, b, acc[t]);
    }
    __syncthreads();
  }

#pragma unroll
  for (int t = 0; t < 4; ++t) {
#pragma unroll
    for (int r = 0; r < 8; ++r) {
      int row = m0 + wave * 16 + r + 8 * half;
      int col = n0 + 16 * t + l15;
      float v = acc[t][r] + bias[col];
      if (GELU) {
        float x3 = v * v * v;
        v = 0.5f * v * (1.f + tanhf(0.7978845608028654f * (v + 0.044715f * x3)));
      }
      if (OUTMODE == 0) {
        ((unsigned short*)outv)[(size_t)row * Nc + col] = f2bf(v);
      } else {
        float* o = (float*)outv;
        size_t p = (size_t)row * Nc + col;
        o[p] = o[p] + v;
      }
    }
  }
}

// ---------------- Flash attention: 1 wave per (b, h, 16-query tile) ----------------
__global__ __launch_bounds__(32)
void k_attn(const unsigned short* __restrict__ qkv, unsigned short* __restrict__ outp) {
  __shared__ unsigned short pL[16 * 32];   // P tile [row][k]
  __shared__ unsigned short vL[64 * 32];   // V tile transposed [dim][key]
  const int lane = threadIdx.x, half = lane >> 4, l15 = lane & 15;
  const int q0 = blockIdx.x * 16, h = blockIdx.y, b = blockIdx.z;
  const unsigned int* q32 = (const unsigned int*)qkv;

  Frag qa[2];
#pragma unroll
  for (int c = 0; c < 2; ++c) {
#pragma unroll
    for (int j = 0; j < 8; ++j) {
      size_t off = (size_t)(b * cN + q0 + l15) * c3D + h * cHD + 32 * c + akk(j, half);
      qa[c].u[j] = q32[off >> 1];
    }
  }

  float mrow[8], lrow[8];
#pragma unroll
  for (int r = 0; r < 8; ++r) { mrow[r] = -1e30f; lrow[r] = 0.f; }
  v8f O[4];
#pragma unroll
  for (int t = 0; t < 4; ++t) O[t] = vzero();

  for (int j0 = 0; j0 < cN; j0 += 32) {
    // load all four K^T fragments (2 key sub-tiles x 2 k-chunks) up front
    Frag kb[2][2];
#pragma unroll
    for (int sub = 0; sub < 2; ++sub) {
#pragma unroll
      for (int c = 0; c < 2; ++c) {
#pragma unroll
        for (int j = 0; j < 8; ++j) {
          size_t off = (size_t)(b * cN + j0 + 16 * sub + l15) * c3D + cD + h * cHD
                     + 32 * c + 16 * half + 2 * j;
          kb[sub][c].u[j] = q32[off >> 1];
        }
      }
    }
    // interleaved WMMA burst: dependent pairs separated by independent WMMA
    v8f s[2];
    s[0] = wmma_bf16(qa[0], kb[0][0], vzero());
    s[1] = wmma_bf16(qa[0], kb[1][0], vzero());
    s[0] = wmma_bf16(qa[1], kb[0][1], s[0]);
    s[1] = wmma_bf16(qa[1], kb[1][1], s[1]);

    float corr[8];
#pragma unroll
    for (int r = 0; r < 8; ++r) {
      float s0 = s[0][r] * 0.125f, s1 = s[1][r] * 0.125f;
      float rm = rmax16(fmaxf(s0, s1));
      float nm = fmaxf(mrow[r], rm);
      corr[r] = __expf(mrow[r] - nm);
      s0 = __expf(s0 - nm);
      s1 = __expf(s1 - nm);
      lrow[r] = lrow[r] * corr[r] + rsum16(s0 + s1);
      mrow[r] = nm;
      s[0][r] = s0; s[1][r] = s1;
    }
#pragma unroll
    for (int t = 0; t < 4; ++t) {
#pragma unroll
      for (int r = 0; r < 8; ++r) O[t][r] *= corr[r];
    }

#pragma unroll
    for (int r = 0; r < 8; ++r) {           // stage P (bf16) in LDS
      pL[(r + 8 * half) * 32 + l15]      = f2bf(s[0][r]);
      pL[(r + 8 * half) * 32 + 16 + l15] = f2bf(s[1][r]);
    }
    {                                        // stage V (32 keys x 64 dims) transposed
      size_t base = (size_t)(b * cN + j0 + lane) * c3D + 2 * cD + h * cHD;
#pragma unroll
      for (int c = 0; c < 32; ++c) {
        unsigned int w = q32[(base + 2 * c) >> 1];
        vL[(2 * c) * 32 + lane]     = (unsigned short)(w & 0xFFFFu);
        vL[(2 * c + 1) * 32 + lane] = (unsigned short)(w >> 16);
      }
    }
    __syncthreads();

    Frag pa;
#pragma unroll
    for (int j = 0; j < 8; ++j)
      pa.u[j] = *(const unsigned int*)&pL[l15 * 32 + akk(j, half)];
#pragma unroll
    for (int t = 0; t < 4; ++t) {
      Frag vb;
#pragma unroll
      for (int j = 0; j < 8; ++j)
        vb.u[j] = *(const unsigned int*)&vL[(16 * t + l15) * 32 + 16 * half + 2 * j];
      O[t] = wmma_bf16(pa, vb, O[t]);
    }
    __syncthreads();
  }

#pragma unroll
  for (int t = 0; t < 4; ++t) {
#pragma unroll
    for (int r = 0; r < 8; ++r) {
      float v = O[t][r] / lrow[r];
      outp[(size_t)(b * cN + q0 + r + 8 * half) * cD + h * cHD + 16 * t + l15] = f2bf(v);
    }
  }
}

// ---------------- pred = x @ proj_w ----------------
__global__ void k_pred(const float* __restrict__ xf, const float* __restrict__ pw,
                       float* __restrict__ pred) {
  int idx = blockIdx.x * blockDim.x + threadIdx.x;
  if (idx >= cM * 3) return;
  int c = idx % 3, bn = idx / 3;
  const float* xr = xf + (size_t)bn * cD;
  float s = 0.f;
  for (int k = 0; k < cD; ++k) s += xr[k] * pw[k * 3 + c];
  pred[idx] = s;
}

// ---------------- FPS: one block per batch ----------------
__global__ __launch_bounds__(256)
void k_fps(const float* __restrict__ pts, float* __restrict__ centers) {
  __shared__ float dist[cP];
  __shared__ float smax[256];
  __shared__ int   sidx[256];
  __shared__ int   sLast;
  int b = blockIdx.x, tid = threadIdx.x;
  for (int i = tid; i < cP; i += 256) dist[i] = 1e10f;
  if (tid == 0) sLast = 0;
  __syncthreads();
  for (int n = 0; n < cN; ++n) {
    int last = sLast;
    const float* lp = pts + (size_t)(b * cP + last) * 3;
    float lx = lp[0], ly = lp[1], lz = lp[2];
    if (tid < 3) centers[((size_t)b * cN + n) * 3 + tid] = lp[tid];
    float bm = -1.f; int bi = 0;
    for (int i = tid; i < cP; i += 256) {
      const float* pp = pts + (size_t)(b * cP + i) * 3;
      float dx = pp[0] - lx, dy = pp[1] - ly, dz = pp[2] - lz;
      float d = dx * dx + dy * dy + dz * dz;
      float nd = fminf(dist[i], d);
      dist[i] = nd;
      if (nd > bm || (nd == bm && i < bi)) { bm = nd; bi = i; }
    }
    smax[tid] = bm; sidx[tid] = bi;
    __syncthreads();
    for (int st = 128; st > 0; st >>= 1) {
      if (tid < st) {
        float ov = smax[tid + st]; int oi = sidx[tid + st];
        if (ov > smax[tid] || (ov == smax[tid] && oi < sidx[tid])) { smax[tid] = ov; sidx[tid] = oi; }
      }
      __syncthreads();
    }
    if (tid == 0) sLast = sidx[0];
    __syncthreads();
  }
}

// ---------------- Chamfer directional min ----------------
__global__ void k_minnn(const float* __restrict__ srcA, const float* __restrict__ srcB,
                        float* __restrict__ mins) {
  int idx = blockIdx.x * blockDim.x + threadIdx.x;
  if (idx >= cB * cN) return;
  int b = idx / cN;
  const float* p = srcA + (size_t)idx * 3;
  float px = p[0], py = p[1], pz = p[2];
  const float* q0 = srcB + (size_t)b * cN * 3;
  float best = 1e30f;
  for (int j = 0; j < cN; ++j) {
    float dx = px - q0[j * 3], dy = py - q0[j * 3 + 1], dz = pz - q0[j * 3 + 2];
    best = fminf(best, dx * dx + dy * dy + dz * dz);
  }
  mins[idx] = sqrtf(best);
}

// ---------------- mean of 5 NN distances (excluding self) ----------------
__global__ void k_meand(const float* __restrict__ pred, float* __restrict__ md) {
  int idx = blockIdx.x * blockDim.x + threadIdx.x;
  if (idx >= cB * cN) return;
  int b = idx / cN;
  const float* p = pred + (size_t)idx * 3;
  float px = p[0], py = p[1], pz = p[2];
  const float* q0 = pred + (size_t)b * cN * 3;
  float t[6];
#pragma unroll
  for (int k = 0; k < 6; ++k) t[k] = 1e30f;
  for (int j = 0; j < cN; ++j) {
    float dx = px - q0[j * 3], dy = py - q0[j * 3 + 1], dz = pz - q0[j * 3 + 2];
    float d = dx * dx + dy * dy + dz * dz;
    if (d < t[5]) {
      int pp = 5;
      while (pp > 0 && t[pp - 1] > d) { t[pp] = t[pp - 1]; --pp; }
      t[pp] = d;
    }
  }
  float s = 0.f;
#pragma unroll
  for (int k = 1; k < 6; ++k) s += sqrtf(t[k]);
  md[idx] = s * 0.2f;
}

__global__ __launch_bounds__(256)
void k_rec(const float* __restrict__ ma, const float* __restrict__ mb, float* __restrict__ out0) {
  __shared__ float sh[256];
  int tid = threadIdx.x;
  float s = 0.f;
  for (int i = tid; i < cB * cN; i += 256) s += ma[i] + mb[i];
  sh[tid] = s; __syncthreads();
  for (int st = 128; st > 0; st >>= 1) { if (tid < st) sh[tid] += sh[tid + st]; __syncthreads(); }
  if (tid == 0) out0[0] = 0.5f * sh[0] / (float)(cB * cN);
}

__global__ __launch_bounds__(256)
void k_klb(const float* __restrict__ md, float* __restrict__ klb) {
  __shared__ float sh[256];
  int b = blockIdx.x, tid = threadIdx.x;
  const float* m = md + (size_t)b * cN;
  float mx = -1e30f;
  for (int i = tid; i < cN; i += 256) mx = fmaxf(mx, m[i]);
  sh[tid] = mx; __syncthreads();
  for (int st = 128; st > 0; st >>= 1) { if (tid < st) sh[tid] = fmaxf(sh[tid], sh[tid + st]); __syncthreads(); }
  mx = sh[0]; __syncthreads();
  float se = 0.f, sm = 0.f;
  for (int i = tid; i < cN; i += 256) { se += __expf(m[i] - mx); sm += m[i]; }
  sh[tid] = se; __syncthreads();
  for (int st = 128; st > 0; st >>= 1) { if (tid < st) sh[tid] += sh[tid + st]; __syncthreads(); }
  float sumexp = sh[0]; __syncthreads();
  sh[tid] = sm; __syncthreads();
  for (int st = 128; st > 0; st >>= 1) { if (tid < st) sh[tid] += sh[tid + st]; __syncthreads(); }
  if (tid == 0)
    klb[b] = -logf((float)cN) - sh[0] / (float)cN + (mx + logf(sumexp));
}

__global__ void k_klf(const float* __restrict__ klb, float* __restrict__ out) {
  if (threadIdx.x == 0) {
    float s = 0.f;
    for (int b = 0; b < cB; ++b) s += klb[b];
    out[1] = s / (float)cB;   // KL_RATIO = 1.0
  }
}

// ---------------- host launcher ----------------
extern "C" void kernel_launch(void* const* d_in, const int* in_sizes, int n_in,
                              void* d_out, int out_size, void* d_ws, size_t ws_size,
                              hipStream_t stream) {
  const float* pts     = (const float*)d_in[0];
  const float* grid    = (const float*)d_in[1];
  const float* embed_w = (const float*)d_in[2];
  const float* proj_w  = (const float*)d_in[3];
  const float* ln1_w   = (const float*)d_in[4];
  const float* ln1_b   = (const float*)d_in[5];
  const float* qkv_w   = (const float*)d_in[6];
  const float* qkv_b   = (const float*)d_in[7];
  const float* attn_w  = (const float*)d_in[8];
  const float* attn_b  = (const float*)d_in[9];
  const float* ln2_w   = (const float*)d_in[10];
  const float* ln2_b   = (const float*)d_in[11];
  const float* mlp_w1  = (const float*)d_in[12];
  const float* mlp_b1  = (const float*)d_in[13];
  const float* mlp_w2  = (const float*)d_in[14];
  const float* mlp_b2  = (const float*)d_in[15];
  float* out = (float*)d_out;

  char* ws = (char*)d_ws;
  size_t off = 0;
  auto take = [&](size_t bytes) -> char* {
    char* p = ws + off;
    off += bytes;
    off = (off + 255) & ~(size_t)255;
    return p;
  };
  float*          xf    = (float*)take((size_t)cM * cD * 4);
  unsigned short* hbf   = (unsigned short*)take((size_t)cM * cD * 2);
  unsigned short* qkvbf = (unsigned short*)take((size_t)cM * c3D * 2);
  unsigned short* aobf  = (unsigned short*)take((size_t)cM * cD * 2);
  unsigned short* hidbf = (unsigned short*)take((size_t)cM * c4D * 2);
  unsigned short* wqkv  = (unsigned short*)take((size_t)cL * cD * c3D * 2);
  unsigned short* wat   = (unsigned short*)take((size_t)cL * cD * cD * 2);
  unsigned short* wm1   = (unsigned short*)take((size_t)cL * cD * c4D * 2);
  unsigned short* wm2   = (unsigned short*)take((size_t)cL * c4D * cD * 2);
  float* pred  = (float*)take((size_t)cB * cN * 3 * 4);
  float* cent  = (float*)take((size_t)cB * cN * 3 * 4);
  float* minA  = (float*)take((size_t)cB * cN * 4);
  float* minB  = (float*)take((size_t)cB * cN * 4);
  float* meand = (float*)take((size_t)cB * cN * 4);
  float* klb   = (float*)take((size_t)cB * 4);

  auto cdiv = [](long a, long b) { return (int)((a + b - 1) / b); };

  // weight conversion f32 -> bf16
  k_cvt<<<cdiv((long)cL * cD * c3D, 256), 256, 0, stream>>>(qkv_w, wqkv, cL * cD * c3D);
  k_cvt<<<cdiv((long)cL * cD * cD, 256), 256, 0, stream>>>(attn_w, wat, cL * cD * cD);
  k_cvt<<<cdiv((long)cL * cD * c4D, 256), 256, 0, stream>>>(mlp_w1, wm1, cL * cD * c4D);
  k_cvt<<<cdiv((long)cL * c4D * cD, 256), 256, 0, stream>>>(mlp_w2, wm2, cL * c4D * cD);

  k_embed<<<cdiv((long)cM * cD, 256), 256, 0, stream>>>(grid, embed_w, xf);

  for (int l = 0; l < cL; ++l) {
    k_layernorm<<<cM / 8, 256, 0, stream>>>(xf, ln1_w + l * cD, ln1_b + l * cD, hbf);
    k_gemm<0, false><<<dim3(c3D / 64, cM / 64), 128, 0, stream>>>(
        hbf, wqkv + (size_t)l * cD * c3D, qkv_b + l * c3D, qkvbf, cM, cD, c3D);
    k_attn<<<dim3(cN / 16, cH, cB), 32, 0, stream>>>(qkvbf, aobf);
    k_gemm<1, false><<<dim3(cD / 64, cM / 64), 128, 0, stream>>>(
        aobf, wat + (size_t)l * cD * cD, attn_b + l * cD, xf, cM, cD, cD);
    k_layernorm<<<cM / 8, 256, 0, stream>>>(xf, ln2_w + l * cD, ln2_b + l * cD, hbf);
    k_gemm<0, true><<<dim3(c4D / 64, cM / 64), 128, 0, stream>>>(
        hbf, wm1 + (size_t)l * cD * c4D, mlp_b1 + l * c4D, hidbf, cM, cD, c4D);
    k_gemm<1, false><<<dim3(cD / 64, cM / 64), 128, 0, stream>>>(
        hidbf, wm2 + (size_t)l * c4D * cD, mlp_b2 + l * cD, xf, cM, c4D, cD);
  }

  k_pred<<<cdiv((long)cM * 3, 256), 256, 0, stream>>>(xf, proj_w, pred);
  k_fps<<<cB, 256, 0, stream>>>(pts, cent);
  k_minnn<<<cdiv(cM, 256), 256, 0, stream>>>(pred, cent, minA);
  k_minnn<<<cdiv(cM, 256), 256, 0, stream>>>(cent, pred, minB);
  k_meand<<<cdiv(cM, 256), 256, 0, stream>>>(pred, meand);
  k_rec<<<1, 256, 0, stream>>>(minA, minB, out);
  k_klb<<<cB, 256, 0, stream>>>(meand, klb);
  k_klf<<<1, 32, 0, stream>>>(klb, out);

  (void)in_sizes; (void)n_in; (void)out_size; (void)ws_size;
}
